// Tokenizer_41197326303537
// MI455X (gfx1250) — compile-verified, matrix operand order checked by reading, not run on hardware
//
#include <hip/hip_runtime.h>
#include <stdint.h>

// ---------- types ----------
typedef __attribute__((ext_vector_type(16))) __bf16 v16bf;
typedef __attribute__((ext_vector_type(8)))  __bf16 v8bf;
typedef __attribute__((ext_vector_type(8)))  float  v8f;
typedef __attribute__((ext_vector_type(4)))  uint32_t v4u;
typedef __attribute__((ext_vector_type(8)))  int      v8i;
typedef __attribute__((ext_vector_type(4)))  int      v4i;

#define B_    256
#define CZ_   512
#define E_    512
#define V_    4096
#define HW_   256
#define N_    (B_ * HW_)          // 65536 spatial positions

// ---------- helpers ----------
__device__ __forceinline__ unsigned short bfbits(float f) {
  union { float f; uint32_t u; } v; v.f = f;
  uint32_t r = v.u + 0x7FFFu + ((v.u >> 16) & 1u);   // RNE
  return (unsigned short)(r >> 16);
}
__device__ __forceinline__ __bf16 f2bf(float f) {
  unsigned short h = bfbits(f);
  union { unsigned short u; __bf16 b; } c; c.u = h; return c.b;
}
__device__ __forceinline__ uint32_t packbf2(float a, float b) {
  return (uint32_t)bfbits(a) | ((uint32_t)bfbits(b) << 16);
}
__device__ __forceinline__ v16bf mk16(v8bf lo, v8bf hi) {
  v16bf r;
#pragma unroll
  for (int i = 0; i < 8; ++i) { r[i] = lo[i]; r[i + 8] = hi[i]; }
  return r;
}
__device__ __forceinline__ v8f vzero8() {
  v8f r;
#pragma unroll
  for (int i = 0; i < 8; ++i) r[i] = 0.0f;
  return r;
}

// ---------- TDM: 2D tile load, 16 rows x 512 bf16 (=256 dwords/row),
// contiguous rows in global, LDS-side padding of 8 dwords per 256-dword row ----------
__device__ __forceinline__ void tdm_load_emb_tile(uint32_t lds_byte_off, const void* gsrc) {
  uint64_t ga = (uint64_t)gsrc;
  v4u g0;
  g0[0] = 1u;                                            // count=1 (valid), no gather
  g0[1] = lds_byte_off;                                  // lds_addr
  g0[2] = (uint32_t)ga;                                  // global_addr[31:0]
  g0[3] = (uint32_t)((ga >> 32) & 0x1FFFFFFu) | (2u << 30); // addr[56:32] | type=2
  v8i g1;
  // data_size=4B(code2), pad_enable, pad_interval=256dw(code7), pad_amount=8dw(code7)
  g1[0] = (int)((2u << 16) | (1u << 20) | (7u << 22) | (7u << 25));
  g1[1] = (int)(256u << 16);          // tensor_dim0 = 256 dwords (lo16 into bits 63:48)
  g1[2] = (int)(4096u << 16);         // tensor_dim0 hi=0 | tensor_dim1=4096 lo16
  g1[3] = (int)(256u << 16);          // tensor_dim1 hi=0 | tile_dim0 = 256 dwords
  g1[4] = (int)(16u);                 // tile_dim1 = 16 rows | tile_dim2 = 0
  g1[5] = (int)(256u);                // tensor_dim0_stride = 256 dwords (lo32)
  g1[6] = 0;                          // stride0 hi | stride1 lo
  g1[7] = 0;
  v4i gz4 = {0, 0, 0, 0};
  v8i gz8 = {0, 0, 0, 0, 0, 0, 0, 0};
  __builtin_amdgcn_tensor_load_to_lds(g0, g1, gz4, gz4, gz8, 0);
}

// =====================================================================
// K0a: fp32 -> bf16 row-major convert
// =====================================================================
__global__ void k_f2bf(const float* __restrict__ in, __bf16* __restrict__ out, int n) {
  int i = blockIdx.x * 256 + threadIdx.x;
  if (i < n) out[i] = f2bf(in[i]);
}

// =====================================================================
// K0b: codebook row norms  cnorm[v] = ||emb_v||^2
// =====================================================================
__global__ void k_cnorm(const float* __restrict__ emb, float* __restrict__ cnorm) {
  __shared__ float red[256];
  int v = blockIdx.x;
  const float* row = emb + (size_t)v * E_;
  float s = 0.f;
  for (int i = threadIdx.x; i < E_; i += 256) { float a = row[i]; s += a * a; }
  red[threadIdx.x] = s;
  __syncthreads();
  for (int st = 128; st > 0; st >>= 1) {
    if (threadIdx.x < st) red[threadIdx.x] += red[threadIdx.x + st];
    __syncthreads();
  }
  if (threadIdx.x == 0) cnorm[v] = red[0];
}

// =====================================================================
// K1: pre-quant GEMM.  z[b,e,hw] = sum_c x[b,c,hw] * pre_w[e,c] + pre_b[e]
//     A = pre_w (16 e-rows x 32 c), B = x tile (32 c x 16 hw) staged via LDS.
//     Writes z fp32 (B,E,H,W) and zf bf16 [n][e] row-major.
// grid: (HW_/16, E_/128, B_)  block: 256 (8 waves; wave w -> e rows w*16..)
// =====================================================================
__global__ __launch_bounds__(256) void k_pre_gemm(
    const float* __restrict__ x, const __bf16* __restrict__ wB,
    const float* __restrict__ pre_b, float* __restrict__ z_out,
    __bf16* __restrict__ zf) {
  __shared__ __align__(16) __bf16 xt[16 * 40];   // [hw][c], row stride 40 halves (80B)
  const int hw0 = blockIdx.x * 16;
  const int b   = blockIdx.z;
  const int tid = threadIdx.x, wid = tid >> 5, lane = tid & 31;
  const int N   = lane & 15;
  const int hiK = (lane >> 4) << 3;              // 0 or 8 (K sub-offset)
  const int hiM = hiK;                           // 0 or 8 (row sub-offset for C)
  const int e0  = blockIdx.y * 128 + wid * 16;

  v8f acc = vzero8();
  for (int ks = 0; ks < 16; ++ks) {
    const int k0 = ks * 32;
    __syncthreads();
    // stage x[c=k0..k0+31][hw0..hw0+15] -> xt[hw][c] as bf16
    for (int j = tid; j < 512; j += 256) {
      int c = j >> 4, i = j & 15;
      float v = x[((size_t)(b * CZ_ + k0 + c)) * HW_ + hw0 + i];
      xt[i * 40 + c] = f2bf(v);
    }
    __syncthreads();
    const __bf16* ap = wB + (size_t)(e0 + N) * CZ_ + k0 + hiK;
    v16bf A = mk16(*(const v8bf*)ap, *(const v8bf*)(ap + 16));
    const __bf16* bp = &xt[N * 40 + hiK];
    v16bf Bf = mk16(*(const v8bf*)bp, *(const v8bf*)(bp + 16));
    acc = __builtin_amdgcn_wmma_f32_16x16x32_bf16(false, A, false, Bf,
                                                  (short)0, acc, false, false);
  }
  // bias + store z fp32 (coalesced: lane = hw)
#pragma unroll
  for (int r = 0; r < 8; ++r) {
    int e = e0 + r + hiM;
    float v = acc[r] + pre_b[e];
    z_out[((size_t)(b * E_ + e)) * HW_ + hw0 + N] = v;
    acc[r] = v;
  }
  // store zf bf16 [n][e], two rows packed per dword
  int n = b * HW_ + hw0 + N;
  uint32_t* zp = (uint32_t*)(zf + (size_t)n * E_ + e0 + hiM);
#pragma unroll
  for (int r = 0; r < 4; ++r) zp[r] = packbf2(acc[2 * r], acc[2 * r + 1]);
}

// =====================================================================
// K2: distance + argmin.  For 16 z-rows per wave (A resident in VGPRs),
//     loop 256 codebook tiles (16 codes x 512) TDM-double-buffered in LDS.
//     dist = cnorm[v] - 2 * (z . emb_v)   (||z||^2 is row-constant)
// grid: N_/128   block: 256
// =====================================================================
#define EROW 528   // LDS row stride in halves: 512 + 16 (TDM pad: 8 dwords)

__global__ __launch_bounds__(256) void k_dist_argmin(
    const __bf16* __restrict__ zf, const __bf16* __restrict__ embb,
    const float* __restrict__ cnorm, int* __restrict__ tokens) {
  __shared__ __align__(128) __bf16 ebuf[2][16 * EROW];   // 2 x 16896 B
  __shared__ float sv[8][16][16];
  __shared__ int   si[8][16][16];
  const int tid = threadIdx.x, wid = tid >> 5, lane = tid & 31;
  const int N = lane & 15, hiK = (lane >> 4) << 3, hi8 = hiK;
  const int n0 = blockIdx.x * 128 + wid * 16;

  // A fragments: this wave's 16 z rows, full K=512, resident
  v16bf A[16];
#pragma unroll
  for (int ks = 0; ks < 16; ++ks) {
    const __bf16* p = zf + (size_t)(n0 + N) * E_ + ks * 32 + hiK;
    A[ks] = mk16(*(const v8bf*)p, *(const v8bf*)(p + 16));
  }

  float bestv[8];
  int   besti[8];
#pragma unroll
  for (int r = 0; r < 8; ++r) { bestv[r] = 3.4e38f; besti[r] = 0; }

  if (wid == 0)
    tdm_load_emb_tile((uint32_t)(uintptr_t)&ebuf[0][0], embb);

  for (int vt = 0; vt < 256; ++vt) {
    const int cur = vt & 1;
    if (wid == 0) {
      if (vt + 1 < 256) {
        tdm_load_emb_tile((uint32_t)(uintptr_t)&ebuf[cur ^ 1][0],
                          embb + (size_t)(vt + 1) * 16 * E_);
        __builtin_amdgcn_s_wait_tensorcnt(1);   // oldest (current tile) done
      } else {
        __builtin_amdgcn_s_wait_tensorcnt(0);   // drain
      }
    }
    __syncthreads();           // publish LDS tile to all waves

    v8f acc = vzero8();
    const __bf16* base = &ebuf[cur][0];
#pragma unroll
    for (int ks = 0; ks < 16; ++ks) {
      const __bf16* bp = base + N * EROW + ks * 32 + hiK;
      v16bf Bf = mk16(*(const v8bf*)bp, *(const v8bf*)(bp + 16));
      acc = __builtin_amdgcn_wmma_f32_16x16x32_bf16(false, A[ks], false, Bf,
                                                    (short)0, acc, false, false);
    }
    float cn = cnorm[vt * 16 + N];
    int   vv = vt * 16 + N;
#pragma unroll
    for (int r = 0; r < 8; ++r) {
      float d = cn - 2.0f * acc[r];
      if (d < bestv[r]) { bestv[r] = d; besti[r] = vv; }
    }
    __syncthreads();           // all readers done before next TDM overwrite
  }

  // reduce across the 16 columns of each row (within-wave via LDS)
#pragma unroll
  for (int r = 0; r < 8; ++r) {
    sv[wid][r + hi8][N] = bestv[r];
    si[wid][r + hi8][N] = besti[r];
  }
  __builtin_amdgcn_wave_barrier();
  if (lane < 16) {
    float bv = sv[wid][lane][0];
    int   bi = si[wid][lane][0];
#pragma unroll
    for (int c = 1; c < 16; ++c) {
      float v = sv[wid][lane][c];
      if (v < bv) { bv = v; bi = si[wid][lane][c]; }
    }
    tokens[n0 + lane] = bi;
  }
}

// =====================================================================
// K3: post-quant GEMM with fused codebook gather.
//     rec[b,c,hw] = sum_e embb[tokens[n]][e] * post_w[c,e] + post_b[c]
//     A = post_w rows (c), B = gathered codebook rows (lane column = hw).
// grid: (HW_/16, CZ_/128, B_)  block: 256
// =====================================================================
__global__ __launch_bounds__(256) void k_post_gemm(
    const __bf16* __restrict__ wB, const __bf16* __restrict__ embb,
    const float* __restrict__ post_b, const int* __restrict__ tokens,
    float* __restrict__ rec) {
  const int hw0 = blockIdx.x * 16;
  const int b   = blockIdx.z;
  const int tid = threadIdx.x, wid = tid >> 5, lane = tid & 31;
  const int N = lane & 15, hiK = (lane >> 4) << 3, hi8 = hiK;
  const int c0 = blockIdx.y * 128 + wid * 16;

  const int t = tokens[b * HW_ + hw0 + N];
  const __bf16* brow = embb + (size_t)t * E_;
  __builtin_prefetch(brow, 0, 3);

  v8f acc = vzero8();
#pragma unroll
  for (int ks = 0; ks < 16; ++ks) {
    const __bf16* ap = wB + (size_t)(c0 + N) * E_ + ks * 32 + hiK;
    v16bf A = mk16(*(const v8bf*)ap, *(const v8bf*)(ap + 16));
    const __bf16* bp = brow + ks * 32 + hiK;
    v16bf Bf = mk16(*(const v8bf*)bp, *(const v8bf*)(bp + 16));
    acc = __builtin_amdgcn_wmma_f32_16x16x32_bf16(false, A, false, Bf,
                                                  (short)0, acc, false, false);
  }
#pragma unroll
  for (int r = 0; r < 8; ++r) {
    int c = c0 + r + hi8;
    rec[((size_t)(b * CZ_ + c)) * HW_ + hw0 + N] = acc[r] + post_b[c];
  }
}

// =====================================================================
// K4: z_q output gather (fp32, layout (B,E,H,W))
// grid: B_*E_  block: 256 (= HW_)
// =====================================================================
__global__ void k_gather_zq(const float* __restrict__ emb,
                            const int* __restrict__ tokens,
                            float* __restrict__ zq) {
  int bid = blockIdx.x;            // b*E_ + e
  int b = bid >> 9, e = bid & 511;
  int hw = threadIdx.x;
  int t = tokens[b * HW_ + hw];
  zq[(size_t)bid * HW_ + hw] = emb[(size_t)t * E_ + e];
}

// =====================================================================
extern "C" void kernel_launch(void* const* d_in, const int* in_sizes, int n_in,
                              void* d_out, int out_size, void* d_ws, size_t ws_size,
                              hipStream_t stream) {
  const float* x      = (const float*)d_in[0];
  const float* pre_w  = (const float*)d_in[1];
  const float* pre_b  = (const float*)d_in[2];
  const float* emb    = (const float*)d_in[3];
  const float* post_w = (const float*)d_in[4];
  const float* post_b = (const float*)d_in[5];

  float* z_out   = (float*)d_out;
  float* zq_out  = z_out + (size_t)B_ * E_ * HW_;
  float* rec_out = zq_out + (size_t)B_ * E_ * HW_;

  // workspace carve-out (256B aligned)
  char* w = (char*)d_ws;
  auto carve = [&](size_t bytes) {
    void* p = (void*)w;
    w += (bytes + 255) & ~(size_t)255;
    return p;
  };
  __bf16* wB1   = (__bf16*)carve((size_t)E_ * CZ_ * 2);   // pre_w bf16
  __bf16* wB3   = (__bf16*)carve((size_t)CZ_ * E_ * 2);   // post_w bf16
  __bf16* embb  = (__bf16*)carve((size_t)V_ * E_ * 2);    // codebook bf16
  float*  cnorm = (float*)carve((size_t)V_ * 4);
  int*    tok   = (int*)carve((size_t)N_ * 4);
  __bf16* zf    = (__bf16*)carve((size_t)N_ * E_ * 2);    // z bf16 [n][e]

  // K0: conversions + norms
  k_f2bf<<<(E_ * CZ_ + 255) / 256, 256, 0, stream>>>(pre_w, wB1, E_ * CZ_);
  k_f2bf<<<(CZ_ * E_ + 255) / 256, 256, 0, stream>>>(post_w, wB3, CZ_ * E_);
  k_f2bf<<<(V_ * E_ + 255) / 256, 256, 0, stream>>>(emb, embb, V_ * E_);
  k_cnorm<<<V_, 256, 0, stream>>>(emb, cnorm);

  // K1: pre-quant GEMM -> z (fp32) + zf (bf16)
  k_pre_gemm<<<dim3(HW_ / 16, E_ / 128, B_), 256, 0, stream>>>(x, wB1, pre_b, z_out, zf);

  // K2: distances + argmin -> tokens
  k_dist_argmin<<<N_ / 128, 256, 0, stream>>>(zf, embb, cnorm, tok);

  // K3: post-quant GEMM (fused gather) -> rec
  k_post_gemm<<<dim3(HW_ / 16, CZ_ / 128, B_), 256, 0, stream>>>(wB3, embb, post_b, tok, rec_out);

  // K4: z_q gather output
  k_gather_zq<<<B_ * E_, HW_, 0, stream>>>(emb, tok, zq_out);

  (void)in_sizes; (void)n_in; (void)out_size; (void)ws_size;
}
